// MKFusion_19567871000740
// MI455X (gfx1250) — compile-verified, matrix operand order checked by reading, not run on hardware
//
#include <hip/hip_runtime.h>

// ---------------------------------------------------------------------------
// MI455X (gfx1250, wave32) CDIM cross-modal attention fusion block.
// All GEMM-shaped work runs on v_wmma_f32_16x16x32_f16 with 4-way register
// blocking and software-pipelined (double-buffered) fragment loads.
// QKV conv input slab is staged to LDS by the Tensor Data Mover.
//
// Layouts:
//   Q/K (s-major)  f16 [B][1024][64]
//   V/DV (c-major) f16 [B][64][1024]
//   logits f32 / mask f16 [B][1024][1024]
//   packed weights f16 [Cout][9*Cin], k = (ky*3+kx)*Cin + cin
// ---------------------------------------------------------------------------

typedef _Float16 v8h  __attribute__((ext_vector_type(8)));
typedef _Float16 v16h __attribute__((ext_vector_type(16)));
typedef float    v8f  __attribute__((ext_vector_type(8)));
typedef float    v4f  __attribute__((ext_vector_type(4)));
typedef unsigned int v4u __attribute__((ext_vector_type(4)));
typedef int      v4i  __attribute__((ext_vector_type(4)));
typedef int      v8i  __attribute__((ext_vector_type(8)));

#define B_  16
#define C_  64
#define H_  128
#define W_  128
#define S_  32
#define S2_ 1024

#if defined(__has_builtin)
#if __has_builtin(__builtin_amdgcn_tensor_load_to_lds)
#define HAVE_TDM 1
#endif
#endif

// WMMA f16 fragment (ISA 7.12.2): lane L holds k in {base..base+7, base+16..23},
// base = (L>>4)*8 -> two 16B loads.
__device__ __forceinline__ v16h frag_ld(const _Float16* p) {
  v8h lo = *(const v8h*)p;
  v8h hi = *(const v8h*)(p + 16);
  return __builtin_shufflevector(lo, hi, 0,1,2,3,4,5,6,7,8,9,10,11,12,13,14,15);
}

__device__ __forceinline__ v8f wmma_f16(v16h a, v16h b, v8f c) {
  return __builtin_amdgcn_wmma_f32_16x16x32_f16(false, a, false, b, (short)0, c,
                                                false, false);
}

__device__ __forceinline__ float cubicK(float t) {
  t = fabsf(t);
  if (t <= 1.f) return ((1.5f * t - 2.5f) * t) * t + 1.f;          // a = -0.5
  if (t < 2.f)  return ((-0.5f * t + 2.5f) * t - 4.f) * t + 2.f;
  return 0.f;
}

// ---------------------------------------------------------------------------
// 1) Weight repack: f32 [Cout][Cin][3][3] -> f16 [Cout][9*Cin]
// ---------------------------------------------------------------------------
__global__ void k_pack_w(const float* __restrict__ w, _Float16* __restrict__ wp,
                         int Cin) {
  int Ktot = Cin * 9;
  int i = blockIdx.x * blockDim.x + threadIdx.x;
  if (i >= 64 * Ktot) return;
  int cout = i / Ktot, k = i % Ktot;
  int tap = k / Cin, cin = k % Cin;
  int ky = tap / 3, kx = tap % 3;
  wp[i] = (_Float16)w[((cout * Cin + cin) * 3 + ky) * 3 + kx];
}

// ---------------------------------------------------------------------------
// 2) Bicubic antialiased downsample 128->32, separable.
// ---------------------------------------------------------------------------
__global__ void k_ds_rows(const float* __restrict__ src, float* __restrict__ tmp) {
  int i = blockIdx.x * blockDim.x + threadIdx.x;           // B*C*32*128
  if (i >= B_ * C_ * 32 * 128) return;
  int x = i & 127, oy = (i >> 7) & 31, c = (i >> 12) & 63, b = i >> 18;
  float center = 4.f * oy + 1.5f;
  float acc = 0.f, wsum = 0.f;
#pragma unroll
  for (int j = 0; j < 16; ++j) {
    int iy = 4 * oy - 6 + j;
    float w = cubicK((iy - center) * 0.25f);
    wsum += w;
    int yc = iy < 0 ? 0 : (iy > 127 ? 127 : iy);
    acc += w * src[(((size_t)b * C_ + c) * H_ + yc) * W_ + x];
  }
  tmp[(((size_t)b * C_ + c) * 32 + oy) * 128 + x] = acc / wsum;
}

__global__ void k_ds_cols(const float* __restrict__ tmp, _Float16* __restrict__ dst) {
  int i = blockIdx.x * blockDim.x + threadIdx.x;           // B*32*32*64
  if (i >= B_ * S_ * S_ * C_) return;
  int c = i & 63, ox = (i >> 6) & 31, oy = (i >> 11) & 31, b = i >> 16;
  float center = 4.f * ox + 1.5f;
  float acc = 0.f, wsum = 0.f;
#pragma unroll
  for (int j = 0; j < 16; ++j) {
    int ix = 4 * ox - 6 + j;
    float w = cubicK((ix - center) * 0.25f);
    wsum += w;
    int xc = ix < 0 ? 0 : (ix > 127 ? 127 : ix);
    acc += w * tmp[(((size_t)b * C_ + c) * 32 + oy) * 128 + xc];
  }
  dst[(((size_t)b * S_ + oy) * S_ + ox) * C_ + c] = (_Float16)(acc / wsum);
}

// ---------------------------------------------------------------------------
// 3) QKV conv (Cin=64, 32x32), input slab staged to LDS via TDM.
//    LDS: row stride 4608B, pixel stride 144B (TDM pad 4dw every 32dw)
//    -> conflict-free ds_load_b128 fragments. Pipelined, 4 accumulators.
// ---------------------------------------------------------------------------
__global__ void k_conv_qkv(const _Float16* __restrict__ in,   // [B][32][32][64]
                           const _Float16* __restrict__ wp,   // [64][576]
                           const float* __restrict__ scale,
                           const float* __restrict__ bias,
                           _Float16* __restrict__ out, int tOut) {
  __shared__ _Float16 Xl[6 * 2304];                  // 6 rows * 4608 bytes
  const int wv = threadIdx.x >> 5, lane = threadIdx.x & 31;
  const int idx = lane & 15, half = lane >> 4, base = half * 8;
  const int b = blockIdx.x >> 3, sblk = blockIdx.x & 7;
  const int y0 = sblk * 4;
  const int ystart = (y0 > 0) ? (y0 - 1) : 0;
  const int yend = (y0 + 4 < 31) ? (y0 + 4) : 31;
  const int nrows = yend - ystart + 1;
  const _Float16* src = in + (((size_t)b * 32 + ystart) * 32) * 64;

#if defined(HAVE_TDM)
  if (threadIdx.x < 32) {                            // wave 0 drives the DMA
    unsigned long long ga = (unsigned long long)(const void*)src;
    v4u g0;                                          // D# group 0 (ISA 8.3)
    g0[0] = 1u;                                      // count=1
    g0[1] = (unsigned)(ystart - (y0 - 1)) * 4608u;   // lds_addr (bytes)
    g0[2] = (unsigned)ga;                            // global_addr lo
    g0[3] = (unsigned)((ga >> 32) & 0x1FFFFFFu) | (2u << 30);  // hi | type=2
    v8i g1;                                          // D# group 1 (ISA 8.4)
    g1[0] = (1 << 16) | (1 << 20) | (4 << 22) | (3 << 25);
    //       data=2B    pad_en      32dw intvl   4dw pad
    g1[1] = (2048 & 0xFFFF) << 16;                   // tensor_dim0 = 2048 elems
    g1[2] = (nrows & 0xFFFF) << 16;                  // tensor_dim1
    g1[3] = (2048 & 0xFFFF) << 16;                   // tile_dim0
    g1[4] = nrows & 0xFFFF;                          // tile_dim1 (tile_dim2=0)
    g1[5] = 2048;                                    // tensor_dim0_stride
    g1[6] = 0; g1[7] = 0;
    v4i g2 = {}; v4i g3 = {};
#if __clang_major__ >= 23
    v8i g4 = {};
    __builtin_amdgcn_tensor_load_to_lds(g0, g1, g2, g3, g4, 0);
#else
    __builtin_amdgcn_tensor_load_to_lds(g0, g1, g2, g3, 0);
#endif
    __builtin_amdgcn_s_wait_tensorcnt(0);
  }
  __syncthreads();
#else
  {                                                  // cooperative padded copy
    int ngroups = nrows * 256;                       // 8-half groups
    for (int p = threadIdx.x; p < ngroups; p += 256) {
      int row = p >> 8, q = p & 255;
      int px = q >> 3, cg = q & 7;
      v8h v = *(const v8h*)(src + (size_t)row * 2048 + q * 8);
      *(v8h*)&Xl[(unsigned)(ystart - (y0 - 1) + row) * 2304u + px * 72u + cg * 8u] = v;
    }
    __syncthreads();
  }
#endif
  // The TDM writes Xl through the DMA engine, invisible to alias analysis:
  // force the compiler to treat LDS as written (else the ds_loads fold away).
  asm volatile("" ::: "memory");

  const int s0 = sblk * 128 + wv * 16;
  const int y = s0 >> 5, x0 = s0 & 31;

  auto ldW = [&](int ch, int ct) -> v16h {
    v16h z = {};
    if (ch >= 18) return z;
    const int tap = ch >> 1, cb = (ch & 1) * 32;
    return frag_ld(wp + (size_t)(ct * 16 + idx) * 576 + tap * 64 + cb + base);
  };
  auto ldX = [&](int ch) -> v16h {
    v16h z = {};
    if (ch >= 18) return z;
    const int tap = ch >> 1, cb = (ch & 1) * 32;
    const int ky = tap / 3, kx = tap % 3;
    const int yy = y + ky - 1, xx = x0 + idx + kx - 1;
    if (yy >= 0 && yy < 32 && xx >= 0 && xx < 32)
      z = frag_ld(&Xl[(unsigned)(yy - y0 + 1) * 2304u + (unsigned)xx * 72u +
                      cb + base]);
    return z;
  };

  v8f acc0 = {}, acc1 = {}, acc2 = {}, acc3 = {};
  v16h bC = ldX(0);
  v16h w0C = ldW(0, 0), w1C = ldW(0, 1), w2C = ldW(0, 2), w3C = ldW(0, 3);
  for (int ch = 0; ch < 18; ++ch) {
    v16h bN = ldX(ch + 1);
    v16h w0N = ldW(ch + 1, 0), w1N = ldW(ch + 1, 1);
    v16h w2N = ldW(ch + 1, 2), w3N = ldW(ch + 1, 3);
    acc0 = wmma_f16(w0C, bC, acc0);
    acc1 = wmma_f16(w1C, bC, acc1);
    acc2 = wmma_f16(w2C, bC, acc2);
    acc3 = wmma_f16(w3C, bC, acc3);
    bC = bN; w0C = w0N; w1C = w1N; w2C = w2N; w3C = w3N;
  }

  const int s = y * 32 + x0 + idx;
  auto emit = [&](v8f a, int ct) {
#pragma unroll
    for (int i = 0; i < 8; ++i) {
      int cout = ct * 16 + i + 8 * half;
      float v = fmaxf(a[i] * scale[cout] + bias[cout], 0.f);
      if (tOut) out[(((size_t)b * 64 + cout) << 10) + s] = (_Float16)v;
      else      out[(((size_t)(b << 10) + s)) * 64 + cout] = (_Float16)v;
    }
  };
  emit(acc0, 0); emit(acc1, 1); emit(acc2, 2); emit(acc3, 3);
}

// ---------------------------------------------------------------------------
// 4) Big 3x3 conv @128x128 (Cin compile-time), fragments from L2,
//    software-pipelined: next-chunk loads issue before current wmmas.
// ---------------------------------------------------------------------------
template <int Cin>
__global__ void k_conv_big(const _Float16* __restrict__ in,   // [B][128][128][Cin]
                           const _Float16* __restrict__ wp,   // [64][9*Cin]
                           const float* __restrict__ scale,
                           const float* __restrict__ bias,
                           _Float16* __restrict__ out16, int outCs, int outCoff,
                           float* __restrict__ out32) {
  constexpr int Ktot = Cin * 9;
  constexpr int CC = Cin / 32;
  constexpr int chunks = CC * 9;
  const int wv = threadIdx.x >> 5, lane = threadIdx.x & 31;
  const int idx = lane & 15, half = lane >> 4, base = half * 8;
  const int b = blockIdx.x >> 7;
  const int y = blockIdx.x & 127;
  const int x0 = wv * 16;
  const _Float16* inb = in + ((size_t)b * H_) * W_ * Cin;
  __builtin_prefetch(inb + ((size_t)y * W_ + x0) * Cin, 0, 0);

  auto ldB = [&](int ch) -> v16h {
    v16h z = {};
    if (ch >= chunks) return z;
    const int tap = ch / CC, cb = (ch % CC) * 32;
    const int ky = tap / 3, kx = tap % 3;
    const int yy = y + ky - 1, xx = x0 + idx + kx - 1;
    if (yy >= 0 && yy < H_ && xx >= 0 && xx < W_)
      z = frag_ld(inb + ((size_t)yy * W_ + xx) * Cin + cb + base);
    return z;
  };
  auto ldA = [&](int ch, int ct) -> v16h {
    v16h z = {};
    if (ch >= chunks) return z;
    const int tap = ch / CC, cb = (ch % CC) * 32;
    return frag_ld(wp + (size_t)(ct * 16 + idx) * Ktot + tap * Cin + cb + base);
  };

  v8f acc0 = {}, acc1 = {}, acc2 = {}, acc3 = {};
  v16h bC = ldB(0);
  v16h a0C = ldA(0, 0), a1C = ldA(0, 1), a2C = ldA(0, 2), a3C = ldA(0, 3);
  for (int ch = 0; ch < chunks; ++ch) {
    v16h bN = ldB(ch + 1);
    v16h a0N = ldA(ch + 1, 0), a1N = ldA(ch + 1, 1);
    v16h a2N = ldA(ch + 1, 2), a3N = ldA(ch + 1, 3);
    acc0 = wmma_f16(a0C, bC, acc0);
    acc1 = wmma_f16(a1C, bC, acc1);
    acc2 = wmma_f16(a2C, bC, acc2);
    acc3 = wmma_f16(a3C, bC, acc3);
    bC = bN; a0C = a0N; a1C = a1N; a2C = a2N; a3C = a3N;
  }

  const int xn = x0 + idx;
  auto emit = [&](v8f a, int ct) {
#pragma unroll
    for (int i = 0; i < 8; ++i) {
      int cout = ct * 16 + i + 8 * half;
      float v = fmaxf(a[i] * scale[cout] + bias[cout], 0.f);
      if (out32)
        out32[(((size_t)b * 64 + cout) * H_ + y) * W_ + xn] = v;
      else
        out16[((size_t)b * H_ * W_ + (size_t)y * W_ + xn) * outCs + outCoff +
              cout] = (_Float16)v;
    }
  };
  emit(acc0, 0); emit(acc1, 1); emit(acc2, 2); emit(acc3, 3);
}

// ---------------------------------------------------------------------------
// 5) DV = RV + IV (both c-major)
// ---------------------------------------------------------------------------
__global__ void k_dv(const _Float16* __restrict__ a, const _Float16* __restrict__ b,
                     _Float16* __restrict__ o, int n) {
  int i = blockIdx.x * blockDim.x + threadIdx.x;
  if (i < n) o[i] = (_Float16)((float)a[i] + (float)b[i]);
}

// ---------------------------------------------------------------------------
// 6) logits[b][s][t] = sum_c Kt[b][s][c]*Qt[b][t][c]. Wave = 16s x 64t strip.
// ---------------------------------------------------------------------------
__global__ void k_logits(const _Float16* __restrict__ Kt,
                         const _Float16* __restrict__ Qt,
                         float* __restrict__ L) {
  const int wv = threadIdx.x >> 5, lane = threadIdx.x & 31;
  const int idx = lane & 15, half = lane >> 4, base = half * 8;
  const int b = blockIdx.x >> 7;                       // 2048 blocks
  const int strip = ((blockIdx.x & 127) << 3) | wv;    // 0..1023
  const int st = strip >> 4, tg = (strip & 15) << 2;   // s-tile, 4 t-tiles
  const _Float16* Kp = Kt + (((size_t)b << 10) << 6);
  const _Float16* Qp = Qt + (((size_t)b << 10) << 6);
  v8f a0 = {}, a1 = {}, a2 = {}, a3 = {};
#pragma unroll
  for (int cb = 0; cb < 64; cb += 32) {
    v16h a = frag_ld(Kp + (size_t)(st * 16 + idx) * 64 + cb + base);
    a0 = wmma_f16(a, frag_ld(Qp + (size_t)((tg + 0) * 16 + idx) * 64 + cb + base), a0);
    a1 = wmma_f16(a, frag_ld(Qp + (size_t)((tg + 1) * 16 + idx) * 64 + cb + base), a1);
    a2 = wmma_f16(a, frag_ld(Qp + (size_t)((tg + 2) * 16 + idx) * 64 + cb + base), a2);
    a3 = wmma_f16(a, frag_ld(Qp + (size_t)((tg + 3) * 16 + idx) * 64 + cb + base), a3);
  }
  float* Lb = L + (((size_t)b << 10) << 10);
  auto emit = [&](v8f a, int j) {
#pragma unroll
    for (int i = 0; i < 8; ++i) {
      int s = st * 16 + i + 8 * half;
      Lb[((size_t)s << 10) + (tg + j) * 16 + idx] = a[i];
    }
  };
  emit(a0, 0); emit(a1, 1); emit(a2, 2); emit(a3, 3);
}

// ---------------------------------------------------------------------------
// 7) softmax over last dim of [16384][1024], f32 -> f16. One wave per row.
// ---------------------------------------------------------------------------
__global__ void k_softmax(const float* __restrict__ L, _Float16* __restrict__ M) {
  const int wv = threadIdx.x >> 5, lane = threadIdx.x & 31;
  const size_t row = (size_t)blockIdx.x * 8 + wv;
  const v4f* p = (const v4f*)(L + (row << 10) + lane * 32);
  float v[32];
#pragma unroll
  for (int j = 0; j < 8; ++j) {
    v4f q = p[j];
    v[4 * j + 0] = q[0]; v[4 * j + 1] = q[1];
    v[4 * j + 2] = q[2]; v[4 * j + 3] = q[3];
  }
  float mx = -3.0e38f;
#pragma unroll
  for (int j = 0; j < 32; ++j) mx = fmaxf(mx, v[j]);
#pragma unroll
  for (int m = 16; m >= 1; m >>= 1) mx = fmaxf(mx, __shfl_xor(mx, m, 32));
  float s = 0.f;
#pragma unroll
  for (int j = 0; j < 32; ++j) { v[j] = __expf(v[j] - mx); s += v[j]; }
#pragma unroll
  for (int m = 16; m >= 1; m >>= 1) s += __shfl_xor(s, m, 32);
  float inv = 1.f / s;
  _Float16* o = M + (row << 10) + lane * 32;
#pragma unroll
  for (int j = 0; j < 32; ++j) o[j] = (_Float16)(v[j] * inv);
}

// ---------------------------------------------------------------------------
// 8) refine[b][s][c] = g * sum_t mask[b][s][t] * Vc[b][c][t], pipelined.
// ---------------------------------------------------------------------------
__global__ void k_refine(const _Float16* __restrict__ M,
                         const _Float16* __restrict__ Vc,
                         const float* __restrict__ gp,
                         float* __restrict__ R) {
  const int wv = threadIdx.x >> 5, lane = threadIdx.x & 31;
  const int idx = lane & 15, half = lane >> 4, base = half * 8;
  const int b = blockIdx.x >> 3, sblk = blockIdx.x & 7;   // 128 blocks
  const int srow0 = sblk * 128 + wv * 16;
  const _Float16* Mb = M + (((size_t)b << 10) << 10) + ((size_t)(srow0 + idx) << 10);
  const _Float16* Vb = Vc + (((size_t)b << 6) << 10);

  auto ldA = [&](int ch) -> v16h {
    v16h z = {};
    if (ch >= 32) return z;
    return frag_ld(Mb + ch * 32 + base);
  };
  auto ldV = [&](int ch, int ct) -> v16h {
    v16h z = {};
    if (ch >= 32) return z;
    return frag_ld(Vb + ((size_t)(ct * 16 + idx) << 10) + ch * 32 + base);
  };

  v8f c0 = {}, c1 = {}, c2 = {}, c3 = {};
  v16h aC = ldA(0);
  v16h v0C = ldV(0, 0), v1C = ldV(0, 1), v2C = ldV(0, 2), v3C = ldV(0, 3);
  for (int ch = 0; ch < 32; ++ch) {
    v16h aN = ldA(ch + 1);
    v16h v0N = ldV(ch + 1, 0), v1N = ldV(ch + 1, 1);
    v16h v2N = ldV(ch + 1, 2), v3N = ldV(ch + 1, 3);
    c0 = wmma_f16(aC, v0C, c0);
    c1 = wmma_f16(aC, v1C, c1);
    c2 = wmma_f16(aC, v2C, c2);
    c3 = wmma_f16(aC, v3C, c3);
    aC = aN; v0C = v0N; v1C = v1N; v2C = v2N; v3C = v3N;
  }
  float g = gp[0];
  auto emit = [&](v8f a, int ct) {
#pragma unroll
    for (int i = 0; i < 8; ++i) {
      int s = srow0 + i + 8 * half;
      R[(((size_t)b << 10) + s) * C_ + ct * 16 + idx] = g * a[i];
    }
  };
  emit(c0, 0); emit(c1, 1); emit(c2, 2); emit(c3, 3);
}

// ---------------------------------------------------------------------------
// 9) bicubic upsample 32->128 + residual -> cat4 NHWC[256] at coff (f16).
// ---------------------------------------------------------------------------
__global__ void k_upadd(const float* __restrict__ R, const float* __restrict__ orig,
                        _Float16* __restrict__ cat, int coff) {
  int i = blockIdx.x * blockDim.x + threadIdx.x;   // B*128*128*4
  if (i >= B_ * H_ * W_ * 4) return;
  int cg = i & 3, X = (i >> 2) & 127, Y = (i >> 9) & 127, b = i >> 16;

  float iny = Y * 0.25f - 0.375f, inx = X * 0.25f - 0.375f;
  float fy = iny - floorf(iny), fx = inx - floorf(inx);
  int iy0 = (int)floorf(iny) - 1, ix0 = (int)floorf(inx) - 1;
  float wy[4] = {cubicK(1.f + fy), cubicK(fy), cubicK(1.f - fy), cubicK(2.f - fy)};
  float wx[4] = {cubicK(1.f + fx), cubicK(fx), cubicK(1.f - fx), cubicK(2.f - fx)};
  int iy[4], ix[4];
#pragma unroll
  for (int j = 0; j < 4; ++j) {
    int a = iy0 + j; iy[j] = a < 0 ? 0 : (a > 31 ? 31 : a);
    int c = ix0 + j; ix[j] = c < 0 ? 0 : (c > 31 ? 31 : c);
  }
  v4f a0 = {}, a1 = {}, a2 = {}, a3 = {};
#pragma unroll
  for (int jy = 0; jy < 4; ++jy)
#pragma unroll
    for (int jx = 0; jx < 4; ++jx) {
      float w = wy[jy] * wx[jx];
      const v4f* rp = (const v4f*)(R + (((size_t)b << 10) + iy[jy] * 32 + ix[jx]) * C_ +
                                   cg * 16);
      a0 += w * rp[0]; a1 += w * rp[1]; a2 += w * rp[2]; a3 += w * rp[3];
    }
  float acc[16];
#pragma unroll
  for (int u = 0; u < 4; ++u) {
    acc[u] = a0[u]; acc[4 + u] = a1[u]; acc[8 + u] = a2[u]; acc[12 + u] = a3[u];
  }
  _Float16* op = cat + ((((size_t)b * H_ + Y) * W_ + X) * 256) + coff + cg * 16;
#pragma unroll
  for (int u = 0; u < 16; ++u) {
    int c = cg * 16 + u;
    float o = orig[(((size_t)b * C_ + c) * H_ + Y) * W_ + X];
    op[u] = (_Float16)(acc[u] + o);
  }
}

// ---------------------------------------------------------------------------
// 10) spatial attention
// ---------------------------------------------------------------------------
__global__ void k_spa_stats(const float* __restrict__ src, float* __restrict__ am) {
  int i = blockIdx.x * blockDim.x + threadIdx.x;   // B*16384
  if (i >= B_ * H_ * W_) return;
  int p = i & 16383, b = i >> 14;
  float mean = 0.f, mx = -3.0e38f;
  for (int c = 0; c < C_; ++c) {
    float v = src[(((size_t)b * C_ + c) << 14) + p];
    mean += v; mx = fmaxf(mx, v);
  }
  am[(((size_t)b * 2 + 0) << 14) + p] = mean * (1.f / C_);
  am[(((size_t)b * 2 + 1) << 14) + p] = mx;
}

__global__ void k_spa_apply(const float* __restrict__ src, const float* __restrict__ am,
                            const float* __restrict__ w, _Float16* __restrict__ secin,
                            int coff) {
  int i = blockIdx.x * blockDim.x + threadIdx.x;   // B*16384
  if (i >= B_ * H_ * W_) return;
  int p = i & 16383, b = i >> 14;
  int X = p & 127, Y = p >> 7;
  float a = 0.f;
#pragma unroll
  for (int ch = 0; ch < 2; ++ch)
#pragma unroll
    for (int ky = 0; ky < 3; ++ky)
#pragma unroll
      for (int kx = 0; kx < 3; ++kx) {
        int yy = Y + ky - 1, xx = X + kx - 1;
        if (yy >= 0 && yy < H_ && xx >= 0 && xx < W_)
          a += w[(ch * 3 + ky) * 3 + kx] *
               am[(((size_t)b * 2 + ch) << 14) + yy * W_ + xx];
      }
  float f = 1.f + 1.f / (1.f + __expf(-a));
  _Float16* op = secin + ((size_t)b * 16384 + p) * 192 + coff;
  for (int c = 0; c < C_; ++c)
    op[c] = (_Float16)(src[(((size_t)b * C_ + c) << 14) + p] * f);
}

// ---------------------------------------------------------------------------
// Host orchestration
// ---------------------------------------------------------------------------
extern "C" void kernel_launch(void* const* d_in, const int* in_sizes, int n_in,
                              void* d_out, int out_size, void* d_ws, size_t ws_size,
                              hipStream_t stream) {
  const float* x = (const float*)d_in[0];
  const float* y = (const float*)d_in[1];
  const float* qkv_w[6], *qkv_s[6], *qkv_b[6];
  for (int i = 0; i < 6; ++i) {
    qkv_w[i] = (const float*)d_in[2 + 3 * i];
    qkv_s[i] = (const float*)d_in[3 + 3 * i];
    qkv_b[i] = (const float*)d_in[4 + 3 * i];
  }
  const float* red_w = (const float*)d_in[20];
  const float* red_s = (const float*)d_in[21];
  const float* red_b = (const float*)d_in[22];
  const float* sec_w = (const float*)d_in[23];
  const float* sec_s = (const float*)d_in[24];
  const float* sec_b = (const float*)d_in[25];
  const float* g[4] = {(const float*)d_in[26], (const float*)d_in[27],
                       (const float*)d_in[28], (const float*)d_in[29]};
  const float* sa_r = (const float*)d_in[30];
  const float* sa_i = (const float*)d_in[31];
  float* out = (float*)d_out;

  char* ws = (char*)d_ws;
  size_t off = 0;
  auto alloc = [&](size_t bytes) -> char* {
    char* p = ws + off;
    off = (off + bytes + 255) & ~(size_t)255;
    return p;
  };

  _Float16* wp_qkv[6];
  for (int i = 0; i < 6; ++i) wp_qkv[i] = (_Float16*)alloc((size_t)64 * 576 * 2);
  _Float16* wp_red = (_Float16*)alloc((size_t)64 * 2304 * 2);
  _Float16* wp_sec = (_Float16*)alloc((size_t)64 * 1728 * 2);
  float* tmpds = (float*)alloc((size_t)B_ * C_ * 32 * 128 * 4);
  _Float16* xr16 = (_Float16*)alloc((size_t)B_ * S2_ * C_ * 2);
  _Float16* yr16 = (_Float16*)alloc((size_t)B_ * S2_ * C_ * 2);
  _Float16* QKV[7];  // Rq Rk Rv Iq Ik Iv DV   (Rv/Iv/DV are c-major)
  for (int i = 0; i < 7; ++i) QKV[i] = (_Float16*)alloc((size_t)B_ * S2_ * C_ * 2);
  float* Lbuf = (float*)alloc((size_t)B_ * S2_ * S2_ * 4);
  _Float16* mask16 = (_Float16*)alloc((size_t)B_ * S2_ * S2_ * 2);
  float* refb = (float*)alloc((size_t)B_ * S2_ * C_ * 4);
  _Float16* cat4 = (_Float16*)alloc((size_t)B_ * H_ * W_ * 256 * 2);
  _Float16* secin = (_Float16*)alloc((size_t)B_ * H_ * W_ * 192 * 2);
  float* am = (float*)alloc((size_t)B_ * 2 * H_ * W_ * 4);

  // ---- 1) pack weights
  for (int i = 0; i < 6; ++i)
    k_pack_w<<<(64 * 576 + 255) / 256, 256, 0, stream>>>(qkv_w[i], wp_qkv[i], 64);
  k_pack_w<<<(64 * 2304 + 255) / 256, 256, 0, stream>>>(red_w, wp_red, 256);
  k_pack_w<<<(64 * 1728 + 255) / 256, 256, 0, stream>>>(sec_w, wp_sec, 192);

  // ---- 2) bicubic downsample
  k_ds_rows<<<16384, 256, 0, stream>>>(x, tmpds);
  k_ds_cols<<<4096, 256, 0, stream>>>(tmpds, xr16);
  k_ds_rows<<<16384, 256, 0, stream>>>(y, tmpds);
  k_ds_cols<<<4096, 256, 0, stream>>>(tmpds, yr16);

  // ---- 3) QKV convs: Q/K s-major, V c-major (tOut=1)
  for (int i = 0; i < 6; ++i) {
    const _Float16* src = (i < 3) ? xr16 : yr16;
    int tOut = (i == 2 || i == 5) ? 1 : 0;
    k_conv_qkv<<<B_ * 8, 256, 0, stream>>>(src, wp_qkv[i], qkv_s[i], qkv_b[i],
                                           QKV[i], tOut);
  }
  k_dv<<<4096, 256, 0, stream>>>(QKV[2], QKV[5], QKV[6], B_ * S2_ * C_);

  // ---- 4) attention branches (Q, K, V, orig, g, coff)
  const _Float16* bq[4] = {QKV[0], QKV[3], QKV[0], QKV[3]};
  const _Float16* bk[4] = {QKV[1], QKV[4], QKV[4], QKV[1]};
  const _Float16* bv[4] = {QKV[6], QKV[6], QKV[2], QKV[5]};
  const float* borig[4] = {x, y, y, x};
  const int bcoff[4] = {0, 64, 128, 192};
  for (int br = 0; br < 4; ++br) {
    k_logits<<<2048, 256, 0, stream>>>(bk[br], bq[br], Lbuf);
    k_softmax<<<2048, 256, 0, stream>>>(Lbuf, mask16);
    k_refine<<<128, 256, 0, stream>>>(mask16, bv[br], g[br], refb);
    k_upadd<<<4096, 256, 0, stream>>>(refb, borig[br], cat4, bcoff[br]);
  }

  // ---- 5) reduction conv 256->64 -> secin channels [0,64)
  k_conv_big<256><<<B_ * 128, 256, 0, stream>>>(cat4, wp_red, red_s, red_b, secin,
                                                192, 0, nullptr);

  // ---- 6) spatial attention -> secin [64,128) (inf) / [128,192) (rgb)
  k_spa_stats<<<1024, 256, 0, stream>>>(y, am);
  k_spa_apply<<<1024, 256, 0, stream>>>(y, am, sa_i, secin, 64);
  k_spa_stats<<<1024, 256, 0, stream>>>(x, am);
  k_spa_apply<<<1024, 256, 0, stream>>>(x, am, sa_r, secin, 128);

  // ---- 7) final conv 192->64, f32 NCHW output
  k_conv_big<192><<<B_ * 128, 256, 0, stream>>>(secin, wp_sec, sec_s, sec_b,
                                                nullptr, 0, 0, out);
}
